// MoeSpaceTimeModel_80264348827721
// MI455X (gfx1250) — matrix operand order repaired and checked by reading, not compiled, not run on hardware
//
#include <hip/hip_runtime.h>

#define NPTS   131072
#define EPSF   1e-8f
#define KP     128      // LDS activation tile row stride, in halfs

typedef _Float16 h8   __attribute__((ext_vector_type(8)));
typedef _Float16 h16  __attribute__((ext_vector_type(16)));
typedef float    f8   __attribute__((ext_vector_type(8)));

// fast reciprocal: single v_rcp_f32 (no IEEE div ladder)
__device__ __forceinline__ float rcp_f(float x){ return __builtin_amdgcn_rcpf(x); }

// silu: gfx1250 hardware tanh (TRANS op, co-issues with VALU):
//   silu(x) = 0.5x + 0.5x*tanh(0.5x)  ->  v_mul + v_tanh_f32 + v_fma
__device__ __forceinline__ float silu_f(float x){
#if __has_builtin(__builtin_amdgcn_tanhf)
  float hx = 0.5f * x;
  float t  = __builtin_amdgcn_tanhf(hx);
  return __builtin_fmaf(hx, t, hx);
#else
  return x * rcp_f(1.0f + __expf(-x));
#endif
}

// 1/(sqrt(msq)+eps) ~= rsq(msq + eps^2): single v_rsq_f32
__device__ __forceinline__ float rrms_f(float msq){
#if __has_builtin(__builtin_amdgcn_rsqf)
  return __builtin_amdgcn_rsqf(msq + 1e-16f);
#else
  return rcp_f(sqrtf(msq) + EPSF);
#endif
}

// 16-lane (half-wave) sum via DPP16 cross-lane adds: no LDS traffic, pure VALU.
// Butterfly order is irrelevant for a sum, so:
//   quad_perm(1,0,3,2)=xor1, quad_perm(2,3,0,1)=xor2,
//   row_half_mirror(lane^7)~xor4 once 4-groups uniform, row_mirror(lane^15)~xor8 once 8-groups uniform.
#if __has_builtin(__builtin_amdgcn_update_dpp)
template<int CTRL>
__device__ __forceinline__ float dpp_add(float v){
  int y = __builtin_amdgcn_update_dpp(0, __float_as_int(v), CTRL, 0xF, 0xF, true);
  return v + __int_as_float(y);
}
__device__ __forceinline__ float row_sum16(float v){
  v = dpp_add<0xB1>(v);   // quad_perm [1,0,3,2]
  v = dpp_add<0x4E>(v);   // quad_perm [2,3,0,1]
  v = dpp_add<0x141>(v);  // row_half_mirror
  v = dpp_add<0x140>(v);  // row_mirror
  return v;
}
#else
__device__ __forceinline__ float row_sum16(float v){
  v += __shfl_xor(v, 1, 32);
  v += __shfl_xor(v, 2, 32);
  v += __shfl_xor(v, 4, 32);
  v += __shfl_xor(v, 8, 32);
  return v;
}
#endif

__device__ __forceinline__ f8 f8zero(){ f8 z = {0.f,0.f,0.f,0.f,0.f,0.f,0.f,0.f}; return z; }

__device__ __forceinline__ h16 cat16(h8 lo, h8 hi){
  return __builtin_shufflevector(lo, hi, 0,1,2,3,4,5,6,7,8,9,10,11,12,13,14,15);
}

// A fragment (16x32 f16, rows = points), CDNA5 WMMA A layout:
// lanes 0-15 : row m=lane,    K = [kb..kb+7]  then [kb+16..kb+23]
// lanes 16-31: row m=lane-16, K = [kb+8..kb+15] then [kb+24..kb+31]
__device__ __forceinline__ h16 frag_A(const _Float16* act, int tile, int kb){
  int lane = threadIdx.x & 31;
  const _Float16* p = act + (tile*16 + (lane & 15)) * KP + kb + ((lane >> 4) << 3);
  h8 lo = *(const h8*)p;
  h8 hi = *(const h8*)(p + 16);
  return cat16(lo, hi);
}

// B fragment (32x16 f16) from transposed weights wT[n][k] (row stride kpad halfs):
// lanes 0-15: col n=lane, K=[kb..kb+7] then [kb+16..kb+23] ; lanes 16-31: col n=lane-16, K=[kb+8..kb+15] then [kb+24..kb+31]
__device__ __forceinline__ h16 frag_B(const _Float16* wT, int kpad, int nb, int kb){
  int lane = threadIdx.x & 31;
  const _Float16* p = wT + (nb*16 + (lane & 15)) * kpad + kb + ((lane >> 4) << 3);
  h8 lo = *(const h8*)p;
  h8 hi = *(const h8*)(p + 16);
  return cat16(lo, hi);
}

__device__ __forceinline__ f8 wmma16(h16 a, h16 b, f8 c){
  return __builtin_amdgcn_wmma_f32_16x16x32_f16(false, a, false, b, (short)0, c, false, false);
}

// ---- bias + [h, silu(h)] concat + rmsnorm(128) row-scale, C-layout -> f16 LDS act tile ----
__device__ __forceinline__ void mid_epilogue(f8 (&acc)[2][4], const float* bias, _Float16* act){
  int lane = threadIdx.x & 31;
  int hb = (lane >> 4) * 8;
  int c0 = lane & 15;
  float b[4];
  #pragma unroll
  for (int nb=0; nb<4; ++nb) b[nb] = bias[nb*16 + c0];
  #pragma unroll
  for (int t=0;t<2;++t){
    #pragma unroll
    for (int r=0;r<8;++r){
      float h[4], s[4]; float ss = 0.f;
      #pragma unroll
      for (int nb=0;nb<4;++nb){
        float v  = acc[t][nb][r] + b[nb];
        float sv = silu_f(v);
        h[nb]=v; s[nb]=sv; ss += v*v + sv*sv;
      }
      ss = row_sum16(ss);                    // per-row (per-point) sum over 128 dims
      float sc = rrms_f(ss * (1.0f/128.0f));
      int row = t*16 + hb + r;
      #pragma unroll
      for (int nb=0;nb<4;++nb){
        int c = nb*16 + c0;
        act[row*KP + c]      = (_Float16)(h[nb]*sc);
        act[row*KP + 64 + c] = (_Float16)(s[nb]*sc);
      }
    }
  }
}

// ---- full SegLU expert stack on one wave's 32-point tile (W1:KPAD1->64, W2:128->64, W3:128->8) ----
template<int KPAD1>
__device__ __forceinline__ void expert_stack(_Float16* act,
    const _Float16* sW1, const _Float16* sW2, const _Float16* sW3,
    const float* b1, const float* b2, const float* b3p,
    const float* wexp_e, f8 (&moe)[2])
{
  int lane = threadIdx.x & 31;
  int hb = (lane >> 4) * 8;
  int c0 = lane & 15;

  f8 acc[2][4];
  #pragma unroll
  for (int t=0;t<2;++t)
    #pragma unroll
    for (int nb=0;nb<4;++nb) acc[t][nb] = f8zero();

  #pragma unroll
  for (int k=0;k<KPAD1;k+=32){
    h16 a0 = frag_A(act, 0, k);
    h16 a1 = frag_A(act, 1, k);
    #pragma unroll
    for (int nb=0;nb<4;++nb){
      h16 b = frag_B(sW1, KPAD1, nb, k);
      acc[0][nb] = wmma16(a0, b, acc[0][nb]);
      acc[1][nb] = wmma16(a1, b, acc[1][nb]);
    }
  }
  mid_epilogue(acc, b1, act);

  #pragma unroll
  for (int t=0;t<2;++t)
    #pragma unroll
    for (int nb=0;nb<4;++nb) acc[t][nb] = f8zero();
  #pragma unroll
  for (int k=0;k<128;k+=32){
    h16 a0 = frag_A(act, 0, k);
    h16 a1 = frag_A(act, 1, k);
    #pragma unroll
    for (int nb=0;nb<4;++nb){
      h16 b = frag_B(sW2, 128, nb, k);
      acc[0][nb] = wmma16(a0, b, acc[0][nb]);
      acc[1][nb] = wmma16(a1, b, acc[1][nb]);
    }
  }
  mid_epilogue(acc, b2, act);

  f8 c3[2] = { f8zero(), f8zero() };
  #pragma unroll
  for (int k=0;k<128;k+=32){
    h16 a0 = frag_A(act, 0, k);
    h16 a1 = frag_A(act, 1, k);
    h16 b  = frag_B(sW3, 128, 0, k);
    c3[0] = wmma16(a0, b, c3[0]);
    c3[1] = wmma16(a1, b, c3[1]);
  }
  float bc = b3p[c0];
  #pragma unroll
  for (int t=0;t<2;++t)
    #pragma unroll
    for (int r=0;r<8;++r){
      int row = t*16 + hb + r;
      moe[t][r] += (c3[t][r] + bc) * wexp_e[row*8];
    }
}

// ---------------- scalar gate SegLU (din=4, inner=8, dout=EOUT) ----------------
struct SegluPtrs { const float *sb,*cb,*g1,*W1,*b1,*g2,*W2,*b2,*g3,*W3,*b3; };

template<int EOUT>
__device__ __forceinline__ void seglu_gate(const SegluPtrs& P, const float x[4], float* lg){
  float f[16]; float ss=0.f;
  float sbv=P.sb[0], cbv=P.cb[0];
  #pragma unroll
  for (int i=0;i<4;++i){
    float xs=x[i]+sbv, xc=x[i]+cbv;
    float v0=__sinf(xs), v1=__sinf(silu_f(xs)), v2=__cosf(xc), v3=__cosf(silu_f(xc));
    f[i]=v0; f[4+i]=v1; f[8+i]=v2; f[12+i]=v3;
    ss += v0*v0+v1*v1+v2*v2+v3*v3;
  }
  float sc=rrms_f(ss*(1.0f/16.0f));
  // fold scale+gain into the inputs once; inner loops are pure fma
  float fn[16];
  #pragma unroll
  for (int k=0;k<16;++k) fn[k] = f[k]*sc*P.g1[k];
  float h1[16]; ss=0.f;
  #pragma unroll
  for (int j=0;j<8;++j){
    float a=P.b1[j];
    #pragma unroll
    for (int k=0;k<16;++k) a = __builtin_fmaf(fn[k], P.W1[k*8+j], a);
    float s=silu_f(a); h1[j]=a; h1[8+j]=s; ss += a*a + s*s;
  }
  sc=rrms_f(ss*(1.0f/16.0f));
  #pragma unroll
  for (int k=0;k<16;++k) fn[k] = h1[k]*sc*P.g2[k];
  float h2[16]; ss=0.f;
  #pragma unroll
  for (int j=0;j<8;++j){
    float a=P.b2[j];
    #pragma unroll
    for (int k=0;k<16;++k) a = __builtin_fmaf(fn[k], P.W2[k*8+j], a);
    float s=silu_f(a); h2[j]=a; h2[8+j]=s; ss += a*a + s*s;
  }
  sc=rrms_f(ss*(1.0f/16.0f));
  #pragma unroll
  for (int k=0;k<16;++k) fn[k] = h2[k]*sc*P.g3[k];
  #pragma unroll
  for (int j=0;j<EOUT;++j){
    float a=P.b3[j];
    #pragma unroll
    for (int k=0;k<16;++k) a = __builtin_fmaf(fn[k], P.W3[k*EOUT+j], a);
    lg[j]=a;
  }
}

// top-2 softmax MoE weights; dense experts [0,ND) weight 1; logit i routes expert ND+i
template<int E, int ND>
__device__ __forceinline__ void moe_weights(const float* lg, float* w){
  #pragma unroll
  for (int i=0;i<E;++i) w[i] = (i<ND) ? 1.0f : 0.0f;
  float v1=-3.4e38f, v2=-3.4e38f; int i1=0, i2=0;
  #pragma unroll
  for (int i=0;i<E;++i){
    float l = lg[i];
    if (l > v1){ v2=v1; i2=i1; v1=l; i1=i; }
    else if (l > v2){ v2=l; i2=i; }
  }
  float e1=1.0f, e2=__expf(v2-v1);
  float inv = rcp_f(e1+e2);
  if (i1 < E-ND) w[ND+i1] += e1*inv;
  if (i2 < E-ND) w[ND+i2] += e2*inv;
}

// =================================== geo stage ===================================
struct GeoArgs { const float* pos; const float* tt; SegluPtrs gate; SegluPtrs ex; float* out; };

__global__ __launch_bounds__(128) void geo_stage(GeoArgs A){
  __shared__ __align__(16) _Float16 sW1[64*32];
  __shared__ __align__(16) _Float16 sW2[64*128];
  __shared__ __align__(16) _Float16 sW3[16*128];
  __shared__ __align__(16) _Float16 sAct[4][32*KP];
  __shared__ float sWexp[128*8];
  __shared__ float sB1[64], sB2[64], sB3[16];

  int tid = threadIdx.x;
  int wave = tid >> 5, lane = tid & 31;
  int p = blockIdx.x*128 + tid;

  float x[4] = { A.pos[p*3+0], A.pos[p*3+1], A.pos[p*3+2], A.tt[p] };

  { float lg[8];
    seglu_gate<8>(A.gate, x, lg);
    moe_weights<8,1>(lg, &sWexp[tid*8]); }

  f8 moe[2] = { f8zero(), f8zero() };
  _Float16* act = sAct[wave];

  for (int e=0;e<8;++e){
    __syncthreads();
    // stage expert weights: f16, transposed [n][k], K-padded, rmsnorm gain folded in
    for (int i=tid;i<16*64;i+=128){ int k=i>>6, n=i&63;
      sW1[n*32+k] = (_Float16)(A.ex.W1[e*1024+i] * A.ex.g1[e*16+k]); }
    for (int i=tid;i<16*64;i+=128){ int k=16+(i>>6), n=i&63; sW1[n*32+k] = (_Float16)0.f; }
    for (int i=tid;i<128*64;i+=128){ int k=i>>6, n=i&63;
      sW2[n*128+k] = (_Float16)(A.ex.W2[e*8192+i] * A.ex.g2[e*128+k]); }
    for (int i=tid;i<128*8;i+=128){ int k=i>>3, n=i&7;
      sW3[n*128+k] = (_Float16)(A.ex.W3[e*1024+i] * A.ex.g3[e*128+k]); }
    for (int i=tid;i<128*8;i+=128){ int n=8+(i>>7), k=i&127; sW3[n*128+k] = (_Float16)0.f; }
    if (tid<64){ sB1[tid]=A.ex.b1[e*64+tid]; sB2[tid]=A.ex.b2[e*64+tid]; }
    if (tid<16) sB3[tid] = (tid<8) ? A.ex.b3[e*8+tid] : 0.f;
    __syncthreads();

    // prefetch next expert's weights into WGP$/L2 (global_prefetch_b8) while we compute
    if (e < 7){
      __builtin_prefetch(A.ex.W2 + (e+1)*8192 + tid*64, 0, 3);
      __builtin_prefetch(A.ex.W1 + (e+1)*1024 + tid*8,  0, 3);
      __builtin_prefetch(A.ex.W3 + (e+1)*1024 + tid*8,  0, 3);
    }

    // layer-0 features (per lane = one point), rms scale folded into f16 write
    float sbv = A.ex.sb[e], cbv = A.ex.cb[e];
    float f[16]; float ss=0.f;
    #pragma unroll
    for (int i=0;i<4;++i){
      float xs = x[i]+sbv, xc = x[i]+cbv;
      float v0=__sinf(xs), v1=__sinf(silu_f(xs)), v2=__cosf(xc), v3=__cosf(silu_f(xc));
      f[i]=v0; f[4+i]=v1; f[8+i]=v2; f[12+i]=v3;
      ss += v0*v0+v1*v1+v2*v2+v3*v3;
    }
    float sc = rrms_f(ss*(1.0f/16.0f));
    #pragma unroll
    for (int i=0;i<16;++i) act[lane*KP+i] = (_Float16)(f[i]*sc);
    #pragma unroll
    for (int i=16;i<32;++i) act[lane*KP+i] = (_Float16)0.f;

    expert_stack<32>(act, sW1, sW2, sW3, sB1, sB2, sB3, &sWexp[wave*32*8 + e], moe);
  }

  int hb = (lane>>4)*8, c0 = lane&15;
  if (c0 < 8){
    #pragma unroll
    for (int t=0;t<2;++t)
      #pragma unroll
      for (int r=0;r<8;++r){
        int row = t*16 + hb + r;
        A.out[(blockIdx.x*128 + wave*32 + row)*8 + c0] = moe[t][r];
      }
  }
}

// =================================== table stage ===================================
struct TabArgs { const float* geo; SegluPtrs gate; const float* tables; float* out; };

__global__ __launch_bounds__(256) void table_stage(TabArgs A){
  int p = blockIdx.x*256 + threadIdx.x;
  const float* g = A.geo + p*8;
  float g1v[4] = { g[0],g[1],g[2],g[3] };
  float lg[6];
  seglu_gate<6>(A.gate, g1v, lg);
  float w[6];
  moe_weights<6,2>(lg, w);
  const int dims[4] = {32,32,32,64};
  int id[4];
  #pragma unroll
  for (int d=0; d<4; ++d){
    int v = (int)(g[4+d] * (float)(dims[d]-1));
    id[d] = min(max(v,0), dims[d]-1);
  }
  size_t off = ((((size_t)id[0]*32 + id[1])*32 + id[2])*64 + id[3]) * 16;
  float acc[16];
  #pragma unroll
  for (int i=0;i<16;++i) acc[i]=0.f;
  #pragma unroll
  for (int e=0;e<6;++e){
    float we = w[e];
    if (we != 0.f){
      const float4* tp = (const float4*)(A.tables + (size_t)e*(size_t)(32*32*32*64*16) + off);
      #pragma unroll
      for (int q=0;q<4;++q){
        float4 v = tp[q];
        acc[q*4+0]+=we*v.x; acc[q*4+1]+=we*v.y; acc[q*4+2]+=we*v.z; acc[q*4+3]+=we*v.w;
      }
    }
  }
  #pragma unroll
  for (int i=0;i<16;++i) A.out[p*16+i]=acc[i];
}

// =================================== render stage ===================================
struct RenArgs { const float* tf; const float* rgw; SegluPtrs ex; const float* origin;
                 const float* aW; const float* aB; const float* cW; const float* cB; float* out; };

__global__ __launch_bounds__(128) void render_stage(RenArgs A){
  __shared__ __align__(16) _Float16 sW1[64*64];
  __shared__ __align__(16) _Float16 sW2[64*128];
  __shared__ __align__(16) _Float16 sW3[16*128];
  __shared__ __align__(16) _Float16 sAct[4][32*KP];
  __shared__ float sWexp[128*8];
  __shared__ float sRend[128*8];
  __shared__ float sB1[64], sB2[64], sB3[16];

  int tid=threadIdx.x, wave=tid>>5, lane=tid&31;
  int p = blockIdx.x*128 + tid;

  float x[16];
  #pragma unroll
  for (int i=0;i<16;++i) x[i]=A.tf[p*16+i];

  { float lg[8];
    #pragma unroll
    for (int j=0;j<8;++j){
      float a=0.f;
      #pragma unroll
      for (int k=0;k<16;++k) a = __builtin_fmaf(x[k], A.rgw[k*8+j], a);
      lg[j]=a;
    }
    moe_weights<8,1>(lg, &sWexp[tid*8]); }

  f8 moe[2] = { f8zero(), f8zero() };
  _Float16* act = sAct[wave];

  for (int e=0;e<8;++e){
    __syncthreads();
    for (int i=tid;i<64*64;i+=128){ int k=i>>6, n=i&63;
      sW1[n*64+k] = (_Float16)(A.ex.W1[e*4096+i]*A.ex.g1[e*64+k]); }
    for (int i=tid;i<128*64;i+=128){ int k=i>>6, n=i&63;
      sW2[n*128+k] = (_Float16)(A.ex.W2[e*8192+i]*A.ex.g2[e*128+k]); }
    for (int i=tid;i<128*8;i+=128){ int k=i>>3, n=i&7;
      sW3[n*128+k] = (_Float16)(A.ex.W3[e*1024+i]*A.ex.g3[e*128+k]); }
    for (int i=tid;i<128*8;i+=128){ int n=8+(i>>7), k=i&127; sW3[n*128+k]=(_Float16)0.f; }
    if (tid<64){ sB1[tid]=A.ex.b1[e*64+tid]; sB2[tid]=A.ex.b2[e*64+tid]; }
    if (tid<16) sB3[tid] = (tid<8)?A.ex.b3[e*8+tid]:0.f;
    __syncthreads();

    if (e < 7){
      __builtin_prefetch(A.ex.W2 + (e+1)*8192 + tid*64, 0, 3);
      __builtin_prefetch(A.ex.W1 + (e+1)*4096 + tid*32, 0, 3);
      __builtin_prefetch(A.ex.W3 + (e+1)*1024 + tid*8,  0, 3);
    }

    float sbv=A.ex.sb[e], cbv=A.ex.cb[e];
    float f[64]; float ss=0.f;
    #pragma unroll
    for (int i=0;i<16;++i){
      float xs=x[i]+sbv, xc=x[i]+cbv;
      float v0=__sinf(xs), v1=__sinf(silu_f(xs)), v2=__cosf(xc), v3=__cosf(silu_f(xc));
      f[i]=v0; f[16+i]=v1; f[32+i]=v2; f[48+i]=v3;
      ss+=v0*v0+v1*v1+v2*v2+v3*v3;
    }
    float sc=rrms_f(ss*(1.0f/64.0f));
    #pragma unroll
    for (int i=0;i<64;++i) act[lane*KP+i]=(_Float16)(f[i]*sc);

    expert_stack<64>(act, sW1, sW2, sW3, sB1, sB2, sB3, &sWexp[wave*32*8+e], moe);
  }

  int hb=(lane>>4)*8, c0=lane&15;
  if (c0<8){
    #pragma unroll
    for (int t=0;t<2;++t)
      #pragma unroll
      for (int r=0;r<8;++r)
        sRend[(wave*32 + t*16 + hb + r)*8 + c0] = moe[t][r];
  }
  __syncthreads();

  // fused head: opacity + color
  float r8[8];
  #pragma unroll
  for (int i=0;i<8;++i) r8[i]=sRend[tid*8+i];
  float op=A.aB[0];
  #pragma unroll
  for (int i=0;i<8;++i) op = __builtin_fmaf(r8[i], A.aW[i], op);
  float cin[12];
  #pragma unroll
  for (int i=0;i<8;++i) cin[i]=r8[i];
  cin[8]=A.origin[p*3+0]; cin[9]=A.origin[p*3+1]; cin[10]=A.origin[p*3+2]; cin[11]=op;
  #pragma unroll
  for (int c=0;c<3;++c){
    float v=A.cB[c];
    #pragma unroll
    for (int k=0;k<12;++k) v = __builtin_fmaf(cin[k], A.cW[k*3+c], v);
    A.out[p*4+c]=v;
  }
  A.out[p*4+3]=op;
}

// =================================== launch ===================================
extern "C" void kernel_launch(void* const* d_in, const int* in_sizes, int n_in,
                              void* d_out, int out_size, void* d_ws, size_t ws_size,
                              hipStream_t stream){
  (void)in_sizes; (void)n_in; (void)out_size; (void)ws_size;
  const float* pos    = (const float*)d_in[0];
  const float* origin = (const float*)d_in[1];
  const float* tt     = (const float*)d_in[2];
  auto seglu = [&](int base){ SegluPtrs s;
    s.sb=(const float*)d_in[base+0]; s.cb=(const float*)d_in[base+1];
    s.g1=(const float*)d_in[base+2]; s.W1=(const float*)d_in[base+3]; s.b1=(const float*)d_in[base+4];
    s.g2=(const float*)d_in[base+5]; s.W2=(const float*)d_in[base+6]; s.b2=(const float*)d_in[base+7];
    s.g3=(const float*)d_in[base+8]; s.W3=(const float*)d_in[base+9]; s.b3=(const float*)d_in[base+10];
    return s; };
  SegluPtrs geo_gate = seglu(3);
  SegluPtrs geo_ex   = seglu(14);
  SegluPtrs tab_gate = seglu(25);
  const float* tables = (const float*)d_in[36];
  const float* rgw    = (const float*)d_in[37];
  SegluPtrs ren_ex    = seglu(38);
  const float* aW=(const float*)d_in[49]; const float* aB=(const float*)d_in[50];
  const float* cW=(const float*)d_in[51]; const float* cB=(const float*)d_in[52];

  float* ws_geo = (float*)d_ws;                    // N x 8
  float* ws_tf  = ws_geo + (size_t)NPTS*8;         // N x 16

  GeoArgs GA{pos, tt, geo_gate, geo_ex, ws_geo};
  geo_stage<<<NPTS/128, 128, 0, stream>>>(GA);

  TabArgs TA{ws_geo, tab_gate, tables, ws_tf};
  table_stage<<<NPTS/256, 256, 0, stream>>>(TA);

  RenArgs RA{ws_tf, rgw, ren_ex, origin, aW, aB, cW, cB, (float*)d_out};
  render_stage<<<NPTS/128, 128, 0, stream>>>(RA);
}
